// DynamicConvLayer_81046032875580
// MI455X (gfx1250) — matrix operand order repaired and checked by reading, not compiled, not run on hardware
//
#include <hip/hip_runtime.h>
#include <hip/hip_bf16.h>

typedef __attribute__((ext_vector_type(16))) _Float16 v16h;
typedef __attribute__((ext_vector_type(8)))  _Float16 v8h;
typedef __attribute__((ext_vector_type(8)))  float    v8f;

#define D_MODEL     128
#define L_SEQ       4096
#define DR          32      // d / reduction
#define KS          3
#define NCHUNK      16
#define CHUNK_TOK   (L_SEQ / NCHUNK)   // 256
#define TOK_PER_BLK 64

// ---------- workspace layout (bytes) ----------
// [0]        wc_h    : 128*128 f16  = 32768 B
// [32768]    wbuf    : 3*128 f32    = 1536 B  (padded to 8192)
// [40960]    partial : 128*16*128 f32 = 1 MB
#define WS_OFF_WCH     0
#define WS_OFF_WBUF    32768
#define WS_OFF_PARTIAL 40960

// ================= kernel 1: Wc f32 -> f16 =================
__global__ void k_wc_to_h(const float* __restrict__ Wc, _Float16* __restrict__ wc_h) {
    int i = blockIdx.x * 256 + threadIdx.x;   // 16384 elems, 64 blocks
    wc_h[i] = (_Float16)Wc[i];
}

// ================= kernel 2: partial sums for mean over L =================
__global__ void k_mean_partial(const float* __restrict__ x, float* __restrict__ partial) {
    const int c     = threadIdx.x;    // 0..127
    const int chunk = blockIdx.x;     // 0..15
    const int b     = blockIdx.y;     // 0..127
    const float* p = x + ((size_t)b * L_SEQ + (size_t)chunk * CHUNK_TOK) * D_MODEL + c;
    float s = 0.f;
    #pragma unroll 8
    for (int t = 0; t < CHUNK_TOK; ++t)
        s += p[(size_t)t * D_MODEL];
    partial[((size_t)b * NCHUNK + chunk) * D_MODEL + c] = s;
}

// ================= kernel 3: tiny MLP + softmax -> w[k][c] =================
__global__ void k_kernelgen(const float* __restrict__ partial,
                            const float* __restrict__ W1, const float* __restrict__ b1,
                            const float* __restrict__ W2, const float* __restrict__ b2,
                            float* __restrict__ wbuf) {
    const int t = threadIdx.x;        // 0..127 : batch row (== channel index, B==d)
    float hacc[DR];
    #pragma unroll
    for (int j = 0; j < DR; ++j) hacc[j] = b1[j];
    // stream over channels: fold mean-reduction directly into the first matmul
    for (int c = 0; c < D_MODEL; ++c) {
        float s = 0.f;
        #pragma unroll
        for (int p = 0; p < NCHUNK; ++p)
            s += partial[((size_t)t * NCHUNK + p) * D_MODEL + c];
        const float m = s * (1.0f / (float)L_SEQ);
        #pragma unroll
        for (int j = 0; j < DR; ++j)
            hacc[j] += m * W1[j * D_MODEL + c];
    }
    // exact GELU
    float h[DR];
    #pragma unroll
    for (int j = 0; j < DR; ++j) {
        const float a = hacc[j];
        h[j] = 0.5f * a * (1.0f + erff(a * 0.70710678118654752f));
    }
    // second matmul + softmax over K=3
    float lg[KS];
    float mx = -1e30f;
    #pragma unroll
    for (int k = 0; k < KS; ++k) {
        float a = b2[k];
        #pragma unroll
        for (int j = 0; j < DR; ++j) a += h[j] * W2[k * DR + j];
        lg[k] = a;
        mx = fmaxf(mx, a);
    }
    float se = 0.f;
    #pragma unroll
    for (int k = 0; k < KS; ++k) { lg[k] = __expf(lg[k] - mx); se += lg[k]; }
    const float inv = 1.0f / se;
    #pragma unroll
    for (int k = 0; k < KS; ++k)
        wbuf[k * D_MODEL + t] = lg[k] * inv;   // [K][c] for coalesced conv reads
}

// ================= kernel 4: fused dynamic conv + pointwise GEMM (WMMA) =================
__global__ __launch_bounds__(256) void k_main(
    const float* __restrict__ x, const float* __restrict__ wbuf,
    const _Float16* __restrict__ wc_h, const float* __restrict__ bc,
    float* __restrict__ out)
{
    __shared__ __align__(32) _Float16 yt[TOK_PER_BLK * D_MODEL];  // 16 KB f16 conv tile
    const int b       = blockIdx.y;
    const int tokbase = blockIdx.x * TOK_PER_BLK;
    const int tid     = threadIdx.x;

    // ---- phase 1: dynamic depthwise 3-tap conv -> f16 tile in LDS ----
    {
        const int c     = tid & 127;         // fixed channel per thread
        const int trow0 = tid >> 7;          // 0..1
        const float w0 = wbuf[0 * D_MODEL + c];
        const float w1 = wbuf[1 * D_MODEL + c];
        const float w2 = wbuf[2 * D_MODEL + c];
        const float* xb = x + (size_t)b * L_SEQ * D_MODEL + c;
        #pragma unroll
        for (int i = 0; i < TOK_PER_BLK / 2; ++i) {
            const int t = trow0 + 2 * i;
            const int g = tokbase + t;
            const float xm = (g > 0)         ? xb[(size_t)(g - 1) * D_MODEL] : 0.f;
            const float xc =                   xb[(size_t)g       * D_MODEL];
            const float xp = (g < L_SEQ - 1) ? xb[(size_t)(g + 1) * D_MODEL] : 0.f;
            yt[t * D_MODEL + c] = (_Float16)(w0 * xm + w1 * xc + w2 * xp);
        }
    }
    __syncthreads();

    // ---- phase 2: out = y @ Wc^T + bc via v_wmma_f32_16x16x32_f16 ----
    const int wave  = tid >> 5;          // 0..7 -> N tile
    const int lane  = tid & 31;
    const int nbase = wave * 16;
    const int lcol  = lane & 15;
    const int lhalf = lane >> 4;

    // B fragments (register-resident across all M tiles):
    // B[k][n] = Wc[n][kb*32 + k]; B k order chosen here (not ISA table order): lane-half holds contiguous 16-K run.
    v16h bfrag[4];
    #pragma unroll
    for (int kb = 0; kb < 4; ++kb) {
        const v16h* bp = (const v16h*)(wc_h + (size_t)(nbase + lcol) * D_MODEL
                                       + kb * 32 + lhalf * 16);
        bfrag[kb] = *bp;
    }
    const float bcv = bc[nbase + lcol];

    #pragma unroll
    for (int mt = 0; mt < TOK_PER_BLK / 16; ++mt) {
        v8f acc = {};
        #pragma unroll
        for (int kb = 0; kb < 4; ++kb) {
            // A k order chosen here: lane-half holds contiguous 16-K run (matches B)
            const v16h* ap = (const v16h*)(yt + (mt * 16 + lcol) * D_MODEL
                                         + kb * 32 + lhalf * 16);
            const v16h a = ap[0];
            acc = __builtin_amdgcn_wmma_f32_16x16x32_f16(
                false, a, false, bfrag[kb], (short)0, acc, false, false);
        }
        // C/D layout: VGPR i -> row M = lhalf*8 + i, col N = lcol
        const int mrow = lhalf * 8;
        float* op = out + ((size_t)b * L_SEQ + tokbase + mt * 16 + mrow) * D_MODEL
                        + nbase + lcol;
        #pragma unroll
        for (int i = 0; i < 8; ++i)
            op[(size_t)i * D_MODEL] = acc[i] + bcv;
    }
}

// ================= launcher =================
extern "C" void kernel_launch(void* const* d_in, const int* in_sizes, int n_in,
                              void* d_out, int out_size, void* d_ws, size_t ws_size,
                              hipStream_t stream) {
    const float* x  = (const float*)d_in[0];
    const float* W1 = (const float*)d_in[1];
    const float* b1 = (const float*)d_in[2];
    const float* W2 = (const float*)d_in[3];
    const float* b2 = (const float*)d_in[4];
    const float* Wc = (const float*)d_in[5];
    const float* bc = (const float*)d_in[6];
    float* out = (float*)d_out;

    char* ws = (char*)d_ws;
    _Float16* wc_h    = (_Float16*)(ws + WS_OFF_WCH);
    float*    wbuf    = (float*)   (ws + WS_OFF_WBUF);
    float*    partial = (float*)   (ws + WS_OFF_PARTIAL);

    // 1) Wc -> f16
    k_wc_to_h<<<dim3(64), dim3(256), 0, stream>>>(Wc, wc_h);
    // 2) mean partials
    k_mean_partial<<<dim3(NCHUNK, D_MODEL), dim3(128), 0, stream>>>(x, partial);
    // 3) kernel-generation MLP + softmax
    k_kernelgen<<<dim3(1), dim3(128), 0, stream>>>(partial, W1, b1, W2, b2, wbuf);
    // 4) fused conv + pointwise GEMM
    k_main<<<dim3(L_SEQ / TOK_PER_BLK, D_MODEL), dim3(256), 0, stream>>>(
        x, wbuf, wc_h, bc, out);
}
